// SL_BNN_20418274525787
// MI455X (gfx1250) — 8-bit matrix operand order repaired and checked by reading, compiled, not run on hardware
//
#include <hip/hip_runtime.h>

typedef __attribute__((ext_vector_type(8))) int v8i;
typedef __attribute__((ext_vector_type(4))) int v4i;

#ifndef __has_builtin
#define __has_builtin(x) 0
#endif

#define ASG __attribute__((address_space(1)))
#define ASL __attribute__((address_space(3)))

#if __has_builtin(__builtin_amdgcn_global_load_async_to_lds_b128)
#define HAVE_ASYNC_LDS 1
#else
#define HAVE_ASYNC_LDS 0
#endif

__device__ __forceinline__ void wait_async_then_barrier() {
#if HAVE_ASYNC_LDS
#if __has_builtin(__builtin_amdgcn_s_wait_asynccnt)
  __builtin_amdgcn_s_wait_asynccnt(0);
#else
  asm volatile("s_wait_asynccnt 0x0" ::: "memory");
#endif
#endif
  __syncthreads();
}

__global__ void binarize_input_kernel(const float* __restrict__ x,
                                      signed char* __restrict__ out, int n) {
  int i = blockIdx.x * blockDim.x + threadIdx.x;
  if (i < n) {
    float v = x[i];
    out[i] = (v > 0.f) ? (signed char)1 : ((v < 0.f) ? (signed char)-1 : (signed char)0);
  }
}

__global__ void pack_w_kernel(const float* __restrict__ w, signed char* __restrict__ w8,
                              int Cout, int K, int CoP, int Kpad) {
  int i = blockIdx.x * blockDim.x + threadIdx.x;
  if (i >= CoP * Kpad) return;
  int co = i / Kpad, k = i % Kpad;
  signed char v = 0;
  if (co < Cout && k < K) {
    float f = w[co * K + k];
    v = (f > 0.f) ? 1 : ((f < 0.f) ? -1 : 0);
  }
  w8[i] = v;
}

template <int CIN, int COUT, int IH, int IW, int OH, int OW, int KC, int NT>
__global__ void conv_wmma_kernel(const signed char* __restrict__ in,
                                 const signed char* __restrict__ w8,
                                 short* __restrict__ yout,
                                 int P) {
  const int K = CIN * 25;
  const int KB = KC * 64;
  __shared__ signed char Bs[16 * KC * 64];

  const int ntile = blockIdx.x % NT;
  const int mtile = (blockIdx.x / NT) * 8 + (threadIdx.x >> 5);
  const int lane  = threadIdx.x & 31;

  {
    const signed char* gsrc = w8 + (long)ntile * (16 * KB);
    for (int i = threadIdx.x * 16; i < 16 * KB; i += 256 * 16) {
#if HAVE_ASYNC_LDS
      __builtin_amdgcn_global_load_async_to_lds_b128(
          (ASG v4i*)(gsrc + i), (ASL v4i*)(Bs + i), 0, 0);
#else
      *(v4i*)(Bs + i) = *(const v4i*)(gsrc + i);
#endif
    }
  }
  wait_async_then_barrier();

  const int lhalf = lane >> 4;
  const int l16   = lane & 15;

  const int p  = mtile * 16 + l16;
  const int n  = p / (OH * OW);
  const int rp = p % (OH * OW);
  const int oh = rp / OW;
  const int ow = rp % OW;
  const int ih0 = oh * 2 - 1;
  const int iw0 = ow * 2 - 1;
  const signed char* inb = in + (long)n * CIN * IH * IW;
  const signed char* brow = Bs + l16 * KB;

  v8i acc = {};
  for (int kc = 0; kc < KC; ++kc) {
    v8i a, b;
#pragma unroll
    for (int v = 0; v < 8; ++v) {
      int kb = kc * 64 + ((v >> 1) << 4) + (lhalf << 3) + ((v & 1) << 2);
      unsigned pack = 0;
#pragma unroll
      for (int byi = 0; byi < 4; ++byi) {
        int k = kb + byi;
        int val = 0;
        if (k < K) {
          int ci = k / 25;
          int r  = k % 25;
          int kh = r / 5;
          int kw = r % 5;
          int ih = ih0 + kh;
          int iw = iw0 + kw;
          if ((unsigned)ih < (unsigned)IH && (unsigned)iw < (unsigned)IW)
            val = inb[(ci * IH + ih) * IW + iw];
        }
        pack |= (unsigned)(unsigned char)(signed char)val << (byi * 8);
      }
      a[v] = (int)pack;
      int kbB = kc * 64 + ((v >> 1) << 4) + (lhalf << 3) + ((v & 1) << 2);
      b[v] = *(const int*)(brow + kbB);
    }
    acc = __builtin_amdgcn_wmma_i32_16x16x64_iu8(true, a, true, b, acc, false, false);
  }

  const int co = ntile * 16 + l16;
  if (co < COUT) {
    short* yrow = yout + (long)co * P;
#pragma unroll
    for (int r = 0; r < 8; ++r) {
      int pm = mtile * 16 + r + lhalf * 8;
      yrow[pm] = (short)acc[r];
    }
  }
}

__global__ void stats_partial_kernel(const short* __restrict__ y, int P, int slices,
                                     float* __restrict__ pbuf) {
  int co = blockIdx.x / slices;
  int s  = blockIdx.x % slices;
  const short* row = y + (long)co * P;
  int start = s * 16384;
  int end = start + 16384; if (end > P) end = P;
  float a = 0.f, b = 0.f;
  for (int i = start + (int)threadIdx.x; i < end; i += 256) {
    float v = (float)row[i]; a += v; b += v * v;
  }
  __shared__ float sa[256], sb[256];
  int t = threadIdx.x;
  sa[t] = a; sb[t] = b;
  __syncthreads();
  for (int off = 128; off > 0; off >>= 1) {
    if (t < off) { sa[t] += sa[t + off]; sb[t] += sb[t + off]; }
    __syncthreads();
  }
  if (t == 0) { pbuf[blockIdx.x * 2] = sa[0]; pbuf[blockIdx.x * 2 + 1] = sb[0]; }
}

__global__ void stats_reduce_kernel(const float* __restrict__ pbuf, int slices,
                                    float* __restrict__ s1, float* __restrict__ s2) {
  int co = blockIdx.x;
  float a = 0.f, b = 0.f;
  for (int s = threadIdx.x; s < slices; s += 256) {
    a += pbuf[(co * slices + s) * 2];
    b += pbuf[(co * slices + s) * 2 + 1];
  }
  __shared__ float sa[256], sb[256];
  int t = threadIdx.x;
  sa[t] = a; sb[t] = b;
  __syncthreads();
  for (int off = 128; off > 0; off >>= 1) {
    if (t < off) { sa[t] += sa[t + off]; sb[t] += sb[t + off]; }
    __syncthreads();
  }
  if (t == 0) { s1[co] = sa[0]; s2[co] = sb[0]; }
}

__global__ void binarize_bn_kernel(const short* __restrict__ y,
                                   const float* __restrict__ s1, const float* __restrict__ s2,
                                   const float* __restrict__ g, const float* __restrict__ be,
                                   signed char* __restrict__ out,
                                   int Cout, int OHW, int P) {
  long idx = (long)blockIdx.x * blockDim.x + threadIdx.x;
  if (idx >= (long)Cout * P) return;
  int co = (int)(idx / P);
  int p  = (int)(idx % P);
  float Pf = (float)P;
  float m  = s1[co] / Pf;
  float var = s2[co] / Pf - m * m;
  float t = ((float)y[(long)co * P + p] - m) * (g[co] / sqrtf(var + 1e-5f)) + be[co];
  int n  = p / OHW;
  int rp = p % OHW;
  out[((long)n * Cout + co) * OHW + rp] = (t > 0.f) ? 1 : 0;
}

__global__ void linear_wmma_kernel(const signed char* __restrict__ a8,
                                   const signed char* __restrict__ b8,
                                   const float* __restrict__ bl,
                                   float* __restrict__ out, int M) {
  const int lane = threadIdx.x & 31;
  const int wave = threadIdx.x >> 5;
  const int wid = blockIdx.x * (blockDim.x >> 5) + wave;
  const int MT = M >> 4;
  const int NT = 23;
  if (wid >= MT * NT) return;
  const int mtile = wid % MT;
  const int ntile = wid / MT;
  const int lhalf = lane >> 4;
  const int l16   = lane & 15;
  const signed char* arow = a8 + (long)(mtile * 16 + l16) * 192;
  const signed char* brow = b8 + (long)(ntile * 16 + l16) * 192;

  v8i acc = {};
#pragma unroll
  for (int kc = 0; kc < 3; ++kc) {
    v8i a, b;
#pragma unroll
    for (int v = 0; v < 8; ++v) {
      int kbA = kc * 64 + ((v >> 1) << 4) + (lhalf << 3) + ((v & 1) << 2);
      int kbB = kc * 64 + ((v >> 1) << 4) + (lhalf << 3) + ((v & 1) << 2);
      a[v] = *(const int*)(arow + kbA);
      b[v] = *(const int*)(brow + kbB);
    }
    acc = __builtin_amdgcn_wmma_i32_16x16x64_iu8(true, a, true, b, acc, false, false);
  }
  int co = ntile * 16 + l16;
  if (co < 360) {
    float bv = bl[co];
#pragma unroll
    for (int r = 0; r < 8; ++r) {
      int m = mtile * 16 + r + lhalf * 8;
      out[(long)m * 360 + co] = (float)acc[r] + bv;
    }
  }
}

extern "C" void kernel_launch(void* const* d_in, const int* in_sizes, int n_in,
                              void* d_out, int out_size, void* d_ws, size_t ws_size,
                              hipStream_t stream) {
  (void)in_sizes; (void)n_in; (void)out_size; (void)ws_size;
  const float* x   = (const float*)d_in[0];
  const float* w1  = (const float*)d_in[1];
  const float* g1  = (const float*)d_in[3];  const float* be1 = (const float*)d_in[4];
  const float* w2  = (const float*)d_in[5];
  const float* g2  = (const float*)d_in[7];  const float* be2 = (const float*)d_in[8];
  const float* w3  = (const float*)d_in[9];
  const float* g3  = (const float*)d_in[11]; const float* be3 = (const float*)d_in[12];
  const float* w4  = (const float*)d_in[13];
  const float* g4  = (const float*)d_in[15]; const float* be4 = (const float*)d_in[16];
  const float* wl  = (const float*)d_in[17]; const float* bl  = (const float*)d_in[18];
  float* out = (float*)d_out;

  const int B = 8192;
  const int P1 = B * 15 * 23;
  const int P2 = B * 7 * 11;
  const int P3 = B * 3 * 5;
  const int P4 = B * 1 * 2;

  char* ws = (char*)d_ws;
  signed char* bin0 = (signed char*)(ws + 0);
  short*       ybuf = (short*)(ws + 75497472UL);
  signed char* bin1 = (signed char*)(ws + 75497472UL + 67829760UL);
  size_t base = 75497472UL + 67829760UL + 33914880UL;
  signed char* w1p = (signed char*)(ws + base);               base += 3072;
  signed char* w2p = (signed char*)(ws + base);               base += 10240;
  signed char* w3p = (signed char*)(ws + base);               base += 30720;
  signed char* w4p = (signed char*)(ws + base);               base += 116736;
  signed char* wlp = (signed char*)(ws + base);               base += 70656;
  float* stats = (float*)(ws + base);                         base += 1536;
  float* pbuf  = (float*)(ws + base);
  float* s1_1 = stats + 0;   float* s2_1 = stats + 12;
  float* s1_2 = stats + 24;  float* s2_2 = stats + 48;
  float* s1_3 = stats + 72;  float* s2_3 = stats + 120;
  float* s1_4 = stats + 168; float* s2_4 = stats + 264;
  signed char* bin2 = bin0;
  signed char* bin3 = bin1;
  signed char* bin4 = bin0;

  const int T = 256;
  { int n = B * 6 * 32 * 48;
    binarize_input_kernel<<<(n + T - 1) / T, T, 0, stream>>>(x, bin0, n); }
  pack_w_kernel<<<(16 * 192 + T - 1) / T, T, 0, stream>>>(w1, w1p, 12, 150, 16, 192);
  pack_w_kernel<<<(32 * 320 + T - 1) / T, T, 0, stream>>>(w2, w2p, 24, 300, 32, 320);
  pack_w_kernel<<<(48 * 640 + T - 1) / T, T, 0, stream>>>(w3, w3p, 48, 600, 48, 640);
  pack_w_kernel<<<(96 * 1216 + T - 1) / T, T, 0, stream>>>(w4, w4p, 96, 1200, 96, 1216);
  pack_w_kernel<<<(368 * 192 + T - 1) / T, T, 0, stream>>>(wl, wlp, 360, 192, 368, 192);

  auto cdiv = [](long a, long b) { return (int)((a + b - 1) / b); };

  { conv_wmma_kernel<6, 12, 32, 48, 15, 23, 3, 1>
        <<<(P1 / 16 / 8) * 1, T, 0, stream>>>(bin0, w1p, ybuf, P1);
    int sl = cdiv(P1, 16384);
    stats_partial_kernel<<<12 * sl, T, 0, stream>>>(ybuf, P1, sl, pbuf);
    stats_reduce_kernel<<<12, T, 0, stream>>>(pbuf, sl, s1_1, s2_1);
    binarize_bn_kernel<<<cdiv((long)12 * P1, T), T, 0, stream>>>(
        ybuf, s1_1, s2_1, g1, be1, bin1, 12, 15 * 23, P1); }

  { conv_wmma_kernel<12, 24, 15, 23, 7, 11, 5, 2>
        <<<(P2 / 16 / 8) * 2, T, 0, stream>>>(bin1, w2p, ybuf, P2);
    int sl = cdiv(P2, 16384);
    stats_partial_kernel<<<24 * sl, T, 0, stream>>>(ybuf, P2, sl, pbuf);
    stats_reduce_kernel<<<24, T, 0, stream>>>(pbuf, sl, s1_2, s2_2);
    binarize_bn_kernel<<<cdiv((long)24 * P2, T), T, 0, stream>>>(
        ybuf, s1_2, s2_2, g2, be2, bin2, 24, 7 * 11, P2); }

  { conv_wmma_kernel<24, 48, 7, 11, 3, 5, 10, 3>
        <<<(P3 / 16 / 8) * 3, T, 0, stream>>>(bin2, w3p, ybuf, P3);
    int sl = cdiv(P3, 16384);
    stats_partial_kernel<<<48 * sl, T, 0, stream>>>(ybuf, P3, sl, pbuf);
    stats_reduce_kernel<<<48, T, 0, stream>>>(pbuf, sl, s1_3, s2_3);
    binarize_bn_kernel<<<cdiv((long)48 * P3, T), T, 0, stream>>>(
        ybuf, s1_3, s2_3, g3, be3, bin3, 48, 3 * 5, P3); }

  { conv_wmma_kernel<48, 96, 3, 5, 1, 2, 19, 6>
        <<<(P4 / 16 / 8) * 6, T, 0, stream>>>(bin3, w4p, ybuf, P4);
    int sl = cdiv(P4, 16384);
    stats_partial_kernel<<<96 * sl, T, 0, stream>>>(ybuf, P4, sl, pbuf);
    stats_reduce_kernel<<<96, T, 0, stream>>>(pbuf, sl, s1_4, s2_4);
    binarize_bn_kernel<<<cdiv((long)96 * P4, T), T, 0, stream>>>(
        ybuf, s1_4, s2_4, g4, be4, bin4, 96, 1 * 2, P4); }

  { int waves = (B / 16) * 23;
    linear_wmma_kernel<<<cdiv(waves, 8), T, 0, stream>>>(bin4, wlp, bl, out, B); }
}
